// DeepGen_u_7310034338077
// MI455X (gfx1250) — compile-verified, matrix operand order checked by reading, not run on hardware
//
#include <hip/hip_runtime.h>

typedef __attribute__((ext_vector_type(16))) _Float16 v16h;
typedef __attribute__((ext_vector_type(8)))  _Float16 v8h;
typedef __attribute__((ext_vector_type(8)))  float    v8f;

#define HDIM 128
#define G_CONST 512
#define KMAX 256
#define ENC_NEG_INF 0x007FFFFFu  // fenc(-inf)

__device__ __forceinline__ unsigned fenc(float x) {
  unsigned u = __float_as_uint(x);
  return (u & 0x80000000u) ? ~u : (u | 0x80000000u);
}
__device__ __forceinline__ float fdec(unsigned e) {
  unsigned u = (e & 0x80000000u) ? (e & 0x7FFFFFFFu) : ~e;
  return __uint_as_float(u);
}

// ---------------------------------------------------------------------------
// WMMA GEMM: C[M,Ncols] = act(A[M,Ktot] @ W[Ncols,Ktot]^T + bias)
// A may be a concat of A1[M,K1] | A2[M,Ktot-K1] along K (A2==null -> K1==Ktot).
// One block (8 waves) owns one 16-row tile_m; the 16 x Ktot A tile is staged
// in LDS as f16 once, then each wave computes one/two 16x16 n-tiles with
// v_wmma_f32_16x16x32_f16. B fragments come straight from global via b128
// loads (W is tiny and L2-resident).
// ---------------------------------------------------------------------------
__global__ __launch_bounds__(256) void k_gemm_wmma(
    const float* __restrict__ A1, const float* __restrict__ A2, int K1, int Ktot,
    const float* __restrict__ W, const float* __restrict__ bias,
    float* __restrict__ C, int Ncols, int ntilesN, int relu) {
  __shared__ _Float16 lds_a[16 * KMAX];
  const int tid = threadIdx.x;
  const int tile_m = blockIdx.x;
  const int K2 = Ktot - K1;
  const int cpr = Ktot >> 2;                 // float4 chunks per row (pow2)
  const int kshift = 31 - __clz(cpr);

  // Cooperative staged copy: A tile (16 x Ktot) -> LDS as f16.
  for (int c = tid; c < 16 * cpr; c += 256) {
    int row = c >> kshift;
    int k = (c & (cpr - 1)) << 2;
    int grow = tile_m * 16 + row;
    float4 v = (k < K1) ? *(const float4*)(A1 + (size_t)grow * K1 + k)
                        : *(const float4*)(A2 + (size_t)grow * K2 + (k - K1));
    _Float16* p = &lds_a[row * Ktot + k];
    p[0] = (_Float16)v.x; p[1] = (_Float16)v.y;
    p[2] = (_Float16)v.z; p[3] = (_Float16)v.w;
  }
  __syncthreads();

  const int wave = tid >> 5;
  const int lane = tid & 31;
  const int idx = lane & 15;                 // M index (A) / N index (B,D)
  const int hh = lane >> 4;                  // K-half selector

  for (int tile_n = wave; tile_n < ntilesN; tile_n += 8) {
    const float* Wr = W + (size_t)(tile_n * 16 + idx) * Ktot;
    v8f acc = {};
    for (int kk = 0; kk < Ktot; kk += 32) {
      // A fragment from LDS: two contiguous 8-half chunks.
      const _Float16* ap = &lds_a[idx * Ktot + kk + hh * 8];
      v8h alo = *(const v8h*)ap;
      v8h ahi = *(const v8h*)(ap + 16);
      // B fragment: two contiguous 8-float chunks -> 4x b128, issued together.
      const float4* wp0 = (const float4*)(Wr + kk + hh * 8);
      const float4* wp1 = (const float4*)(Wr + kk + 16 + hh * 8);
      float4 b0 = wp0[0];
      float4 b1 = wp0[1];
      float4 b2 = wp1[0];
      float4 b3 = wp1[1];
      __builtin_prefetch(Wr + kk + 64, 0, 0);
      v16h a = __builtin_shufflevector(alo, ahi, 0, 1, 2, 3, 4, 5, 6, 7, 8, 9,
                                       10, 11, 12, 13, 14, 15);
      v16h b;
      b[0] = (_Float16)b0.x;  b[1] = (_Float16)b0.y;
      b[2] = (_Float16)b0.z;  b[3] = (_Float16)b0.w;
      b[4] = (_Float16)b1.x;  b[5] = (_Float16)b1.y;
      b[6] = (_Float16)b1.z;  b[7] = (_Float16)b1.w;
      b[8] = (_Float16)b2.x;  b[9] = (_Float16)b2.y;
      b[10] = (_Float16)b2.z; b[11] = (_Float16)b2.w;
      b[12] = (_Float16)b3.x; b[13] = (_Float16)b3.y;
      b[14] = (_Float16)b3.z; b[15] = (_Float16)b3.w;
      acc = __builtin_amdgcn_wmma_f32_16x16x32_f16(false, a, false, b, (short)0,
                                                   acc, false, false);
    }
    float bval = bias ? bias[tile_n * 16 + idx] : 0.0f;
#pragma unroll
    for (int r = 0; r < 8; ++r) {            // D: VGPR r -> row r+8*hh, col idx
      float v = acc[r] + bval;
      if (relu) v = fmaxf(v, 0.0f);
      C[(size_t)(tile_m * 16 + r + 8 * hh) * Ncols + tile_n * 16 + idx] = v;
    }
  }
}

// ---------------------------------------------------------------------------
// Fill (bit-pattern) helper
// ---------------------------------------------------------------------------
__global__ void k_fill_u32(unsigned* __restrict__ p, unsigned v, size_t n) {
  size_t i = (size_t)blockIdx.x * blockDim.x + threadIdx.x;
  size_t stride = (size_t)gridDim.x * blockDim.x;
  for (; i < n; i += stride) p[i] = v;
}

// ---------------------------------------------------------------------------
// Per-node double dot: o1[n]=dot(nf[n],a1), o2[n]=dot(nf[n],a2). Wave per node.
// ---------------------------------------------------------------------------
__global__ void k_node_dot2(const float* __restrict__ nf, const float* __restrict__ a1,
                            const float* __restrict__ a2, float* __restrict__ o1,
                            float* __restrict__ o2, int N) {
  int wid = (blockIdx.x * blockDim.x + threadIdx.x) >> 5;
  int lane = threadIdx.x & 31;
  if (wid >= N) return;
  float4 v  = ((const float4*)(nf + (size_t)wid * HDIM))[lane];
  float4 w1 = ((const float4*)a1)[lane];
  float4 w2 = ((const float4*)a2)[lane];
  float s1 = v.x * w1.x + v.y * w1.y + v.z * w1.z + v.w * w1.w;
  float s2 = v.x * w2.x + v.y * w2.y + v.z * w2.z + v.w * w2.w;
  for (int off = 16; off > 0; off >>= 1) {
    s1 += __shfl_down(s1, off);
    s2 += __shfl_down(s2, off);
  }
  if (lane == 0) { o1[wid] = s1; o2[wid] = s2; }
}

// Per-row dot of two [N,H] matrices. Wave per node.
__global__ void k_row_dot(const float* __restrict__ X, const float* __restrict__ Y,
                          float* __restrict__ out, int N) {
  int wid = (blockIdx.x * blockDim.x + threadIdx.x) >> 5;
  int lane = threadIdx.x & 31;
  if (wid >= N) return;
  float4 a = ((const float4*)(X + (size_t)wid * HDIM))[lane];
  float4 b = ((const float4*)(Y + (size_t)wid * HDIM))[lane];
  float s = a.x * b.x + a.y * b.y + a.z * b.z + a.w * b.w;
  for (int off = 16; off > 0; off >>= 1) s += __shfl_down(s, off);
  if (lane == 0) out[wid] = s;
}

// ---------------------------------------------------------------------------
// Edge phase (segment softmax over dst + weighted aggregation)
// ---------------------------------------------------------------------------
__global__ void k_edge_logit_max(const int* __restrict__ src, const int* __restrict__ dst,
                                 const float* __restrict__ s1, const float* __restrict__ s2,
                                 float* __restrict__ ef, unsigned* __restrict__ nmax, int E) {
  int e = blockIdx.x * blockDim.x + threadIdx.x;
  if (e >= E) return;
  float l = s1[src[e]] + s2[dst[e]];
  ef[e] = l;
  atomicMax(nmax + dst[e], fenc(l));
}

__global__ void k_edge_expsum(const int* __restrict__ dst, float* __restrict__ ef,
                              const unsigned* __restrict__ nmax, float* __restrict__ nsum,
                              int E) {
  int e = blockIdx.x * blockDim.x + threadIdx.x;
  if (e >= E) return;
  int d = dst[e];
  float w = __expf(ef[e] - fdec(nmax[d]));
  ef[e] = w;
  atomicAdd(nsum + d, w);
}

// Wave per edge: agg[dst] += (alpha/nsum[dst]) * nf[src]
__global__ void k_edge_aggregate(const int* __restrict__ src, const int* __restrict__ dst,
                                 const float* __restrict__ alpha, const float* __restrict__ nsum,
                                 const float* __restrict__ nf, float* __restrict__ agg, int E) {
  int wid = (blockIdx.x * blockDim.x + threadIdx.x) >> 5;
  int lane = threadIdx.x & 31;
  if (wid >= E) return;
  int s = src[wid], d = dst[wid];
  float c = alpha[wid] / nsum[d];
  float4 v = ((const float4*)(nf + (size_t)s * HDIM))[lane];
  float* drow = agg + (size_t)d * HDIM + lane * 4;
  atomicAdd(drow + 0, c * v.x);
  atomicAdd(drow + 1, c * v.y);
  atomicAdd(drow + 2, c * v.z);
  atomicAdd(drow + 3, c * v.w);
}

// ---------------------------------------------------------------------------
// Graph pooling helpers
// ---------------------------------------------------------------------------
__global__ void k_node_max(const int* __restrict__ batch, const float* __restrict__ l,
                           unsigned* __restrict__ gmax, int N) {
  int n = blockIdx.x * blockDim.x + threadIdx.x;
  if (n >= N) return;
  atomicMax(gmax + batch[n], fenc(l[n]));
}

__global__ void k_node_expsum(const int* __restrict__ batch, const float* __restrict__ l,
                              const unsigned* __restrict__ gmax, float* __restrict__ gsum,
                              float* __restrict__ w, int N) {
  int n = blockIdx.x * blockDim.x + threadIdx.x;
  if (n >= N) return;
  int g = batch[n];
  float e = __expf(l[n] - fdec(gmax[g]));
  w[n] = e;
  atomicAdd(gsum + g, e);
}

__global__ void k_scale_rows(const float* __restrict__ nf, const float* __restrict__ w,
                             const float* __restrict__ gsum, const int* __restrict__ batch,
                             float* __restrict__ h, int N) {
  int t = blockIdx.x * blockDim.x + threadIdx.x;
  int n = t >> 5, lane = t & 31;
  if (n >= N) return;
  float c = w[n] / gsum[batch[n]];
  float4 v = ((const float4*)(nf + (size_t)n * HDIM))[lane];
  float4 o = {c * v.x, c * v.y, c * v.z, c * v.w};
  ((float4*)(h + (size_t)n * HDIM))[lane] = o;
}

__global__ void k_gf_accum(const int* __restrict__ batch, const float* __restrict__ h,
                           float* __restrict__ gf, int N) {
  int t = blockIdx.x * blockDim.x + threadIdx.x;
  int n = t >> 5, lane = t & 31;
  if (n >= N) return;
  int g = batch[n];
  float4 v = ((const float4*)(h + (size_t)n * HDIM))[lane];
  float* grow = gf + (size_t)g * HDIM + lane * 4;
  atomicAdd(grow + 0, v.x);
  atomicAdd(grow + 1, v.y);
  atomicAdd(grow + 2, v.z);
  atomicAdd(grow + 3, v.w);
}

// ---------------------------------------------------------------------------
// BatchNorm (training): column sums / sums-of-squares, then fused norm+ReLU
// ---------------------------------------------------------------------------
__global__ void k_col_stats(const float* __restrict__ X, float* __restrict__ colsum,
                            float* __restrict__ colsq, int R, int rowsPerBlock) {
  int c = threadIdx.x;  // HDIM threads
  int r0 = blockIdx.x * rowsPerBlock;
  int r1 = min(r0 + rowsPerBlock, R);
  float s = 0.0f, q = 0.0f;
  for (int r = r0; r < r1; ++r) {
    float v = X[(size_t)r * HDIM + c];
    s += v;
    q += v * v;
  }
  atomicAdd(colsum + c, s);
  atomicAdd(colsq + c, q);
}

__global__ void k_bn_relu(float* __restrict__ X, const float* __restrict__ colsum,
                          const float* __restrict__ colsq, const float* __restrict__ g,
                          const float* __restrict__ b, int R) {
  size_t i = (size_t)blockIdx.x * blockDim.x + threadIdx.x;
  if (i >= (size_t)R * HDIM) return;
  int c = (int)(i & (HDIM - 1));
  float invR = 1.0f / (float)R;
  float mu = colsum[c] * invR;
  float var = colsq[c] * invR - mu * mu;
  float v = (X[i] - mu) * rsqrtf(var + 1e-5f) * g[c] + b[c];
  X[i] = fmaxf(v, 0.0f);
}

// ---------------------------------------------------------------------------
extern "C" void kernel_launch(void* const* d_in, const int* in_sizes, int n_in,
                              void* d_out, int out_size, void* d_ws, size_t ws_size,
                              hipStream_t stream) {
  const float* x         = (const float*)d_in[0];
  const int*   ei        = (const int*)d_in[1];
  const float* edge_attr = (const float*)d_in[2];
  const int*   batch     = (const int*)d_in[3];
  const float* Wi   = (const float*)d_in[5];
  const float* bi   = (const float*)d_in[6];
  const float* gatW = (const float*)d_in[7];
  const float* gatA = (const float*)d_in[8];
  const float* nfW1 = (const float*)d_in[9];
  const float* nfb1 = (const float*)d_in[10];
  const float* nf_g = (const float*)d_in[11];
  const float* nf_b = (const float*)d_in[12];
  const float* nfW2 = (const float*)d_in[13];
  const float* nfb2 = (const float*)d_in[14];
  const float* gfW1 = (const float*)d_in[15];
  const float* gfb1 = (const float*)d_in[16];
  const float* gf_g = (const float*)d_in[17];
  const float* gf_b = (const float*)d_in[18];
  const float* gfW2 = (const float*)d_in[19];
  const float* gfb2 = (const float*)d_in[20];

  const int N = in_sizes[3];
  const int E = in_sizes[1] / 2;
  const int G = G_CONST;
  const int DEPTH = 3;
  const int NF_IN = in_sizes[0] / N;   // 64
  const int NF_OUT = 64;
  const int EDIM = 16;
  const int* src = ei;
  const int* dst = ei + E;

  // Workspace layout (floats)
  float* ws = (float*)d_ws;
  size_t o = 0;
  float* nf0 = ws + o;  o += (size_t)N * HDIM;
  float* nf1 = ws + o;  o += (size_t)N * HDIM;
  float* agg = ws + o;  o += (size_t)N * HDIM;   // also pooling h
  float* ef  = ws + o;  o += (size_t)E;
  unsigned* nmax = (unsigned*)(ws + o); o += (size_t)N;
  float* nsum = ws + o; o += (size_t)N;
  float* nw   = ws + o; o += (size_t)N;          // s1 in GAT phase, w in pooling
  float* s2   = ws + o; o += (size_t)N;          // s2 / pooling logits
  unsigned* gmax = (unsigned*)(ws + o); o += (size_t)G;
  float* gsum = ws + o; o += (size_t)G;
  float* gf   = ws + o; o += (size_t)G * HDIM;
  float* gtmp = ws + o; o += (size_t)G * HDIM;
  float* colsum = ws + o; o += HDIM;
  float* colsq  = ws + o; o += HDIM;

  auto fill = [&](void* p, unsigned v, size_t n) {
    int blocks = (int)((n + 255) / 256);
    if (blocks > 4096) blocks = 4096;
    k_fill_u32<<<blocks, 256, 0, stream>>>((unsigned*)p, v, n);
  };
  auto gemm = [&](const float* A1, const float* A2, int K1, int Ktot, const float* W,
                  const float* bias, float* C, int M, int Ncols, int relu) {
    k_gemm_wmma<<<M / 16, 256, 0, stream>>>(A1, A2, K1, Ktot, W, bias, C, Ncols,
                                            Ncols / 16, relu);
  };

  const int tpbE = (E + 255) / 256;          // thread-per-edge
  const int wpbE = (E * 32 + 255) / 256;     // wave-per-edge
  const int tpbN = (N + 255) / 256;          // thread-per-node
  const int wpbN = (N * 32 + 255) / 256;     // wave-per-node

  // 1) input projection: nf = x @ Wi^T + bi
  gemm(x, nullptr, NF_IN, NF_IN, Wi, bi, nf0, N, HDIM, 0);

  // 2) GAT layers
  float* nf_cur = nf0;
  float* nf_nxt = nf1;
  for (int i = 0; i < DEPTH; ++i) {
    fill(agg, 0u, (size_t)N * HDIM);
    fill(nmax, ENC_NEG_INF, (size_t)N);
    fill(nsum, 0u, (size_t)N);
    k_node_dot2<<<wpbN, 256, 0, stream>>>(nf_cur, gatA + (size_t)i * 2 * HDIM,
                                          gatA + (size_t)i * 2 * HDIM + HDIM, nw, s2, N);
    k_edge_logit_max<<<tpbE, 256, 0, stream>>>(src, dst, nw, s2, ef, nmax, E);
    k_edge_expsum<<<tpbE, 256, 0, stream>>>(dst, ef, nmax, nsum, E);
    k_edge_aggregate<<<wpbE, 256, 0, stream>>>(src, dst, ef, nsum, nf_cur, agg, E);
    gemm(nf_cur, agg, HDIM, 2 * HDIM, gatW + (size_t)i * HDIM * 2 * HDIM, nullptr,
         nf_nxt, N, HDIM, 1);
    float* t = nf_cur; nf_cur = nf_nxt; nf_nxt = t;
  }

  // 3) 3-iteration attention pooling over graphs
  float* h = agg;
  hipMemcpyAsync(h, nf_cur, (size_t)N * HDIM * sizeof(float),
                 hipMemcpyDeviceToDevice, stream);
  for (int it = 0; it < 3; ++it) {
    fill(gmax, ENC_NEG_INF, (size_t)G);
    fill(gsum, 0u, (size_t)G);
    k_row_dot<<<wpbN, 256, 0, stream>>>(h, nf_cur, s2, N);
    k_node_max<<<tpbN, 256, 0, stream>>>(batch, s2, gmax, N);
    k_node_expsum<<<tpbN, 256, 0, stream>>>(batch, s2, gmax, gsum, nw, N);
    k_scale_rows<<<wpbN, 256, 0, stream>>>(nf_cur, nw, gsum, batch, h, N);
  }
  fill(gf, 0u, (size_t)G * HDIM);
  k_gf_accum<<<wpbN, 256, 0, stream>>>(batch, h, gf, N);

  // 4) node head: relu(BN(nf@nfW1^T + nfb1)) @ nfW2^T + nfb2 -> d_out[0 : N*64]
  float* t1 = nf_nxt;  // free ping buffer
  gemm(nf_cur, nullptr, HDIM, HDIM, nfW1, nfb1, t1, N, HDIM, 0);
  fill(colsum, 0u, 2 * HDIM);  // colsum + colsq contiguous
  k_col_stats<<<(N + 127) / 128, HDIM, 0, stream>>>(t1, colsum, colsq, N, 128);
  k_bn_relu<<<(int)(((size_t)N * HDIM + 255) / 256), 256, 0, stream>>>(
      t1, colsum, colsq, nf_g, nf_b, N);
  float* out_nf = (float*)d_out;
  gemm(t1, nullptr, HDIM, HDIM, nfW2, nfb2, out_nf, N, NF_OUT, 0);

  // 5) edge_attr passthrough
  float* out_ea = out_nf + (size_t)N * NF_OUT;
  hipMemcpyAsync(out_ea, edge_attr, (size_t)E * EDIM * sizeof(float),
                 hipMemcpyDeviceToDevice, stream);

  // 6) graph head: relu(BN(gf@gfW1^T + gfb1)) @ gfW2^T + gfb2
  gemm(gf, nullptr, HDIM, HDIM, gfW1, gfb1, gtmp, G, HDIM, 0);
  fill(colsum, 0u, 2 * HDIM);
  k_col_stats<<<(G + 127) / 128, HDIM, 0, stream>>>(gtmp, colsum, colsq, G, 128);
  k_bn_relu<<<(int)(((size_t)G * HDIM + 255) / 256), 256, 0, stream>>>(
      gtmp, colsum, colsq, gf_g, gf_b, G);
  float* out_gf = out_ea + (size_t)E * EDIM;
  gemm(gtmp, nullptr, HDIM, HDIM, gfW2, gfb2, out_gf, G, HDIM, 0);
}
